// SA2D_62680752718501
// MI455X (gfx1250) — matrix operand order repaired and checked by reading, not compiled, not run on hardware
//
#include <hip/hip_runtime.h>
#include <hip/hip_bf16.h>

// ---------------------------------------------------------------------------
// Local self-attention (SASA-style), B=8 C=64 H=W=128, 3x3 window, pad 1.
// Phase 1: fused QKV 1x1-conv GEMM via v_wmma_f32_16x16x32_bf16.
// Phase 2: per-pixel 9-way window attention with K/V tiles staged to LDS via
//          gfx1250 async global->LDS loads (guarded; falls back to float4).
// ---------------------------------------------------------------------------

typedef __attribute__((ext_vector_type(16))) __bf16 v16bf;
typedef __attribute__((ext_vector_type(8)))  __bf16 v8bf;
typedef __attribute__((ext_vector_type(8)))  float  v8f;
typedef __attribute__((ext_vector_type(4)))  int    v4i;

#define BATCH 8
#define CH    64
#define HH    128
#define WW    128
#define HP    130
#define WP    130

#define AS1 __attribute__((address_space(1)))
#define AS3 __attribute__((address_space(3)))

#if __has_builtin(__builtin_amdgcn_global_load_async_to_lds_b128) && \
    __has_builtin(__builtin_amdgcn_s_wait_asynccnt)
#define SA2D_ASYNC_LDS 1
#else
#define SA2D_ASYNC_LDS 0
#endif

// ---------------------------------------------------------------------------
// Kernel 0: zero the 1-pixel pad border of the padded K / V buffers.
// ---------------------------------------------------------------------------
__global__ __launch_bounds__(256) void SA2D_zero_pad(float* __restrict__ Kp,
                                                     float* __restrict__ Vp) {
  int idx = blockIdx.x * blockDim.x + threadIdx.x;      // over B*HP*WP
  if (idx >= BATCH * HP * WP) return;
  int p  = idx % (HP * WP);
  int hh = p / WP, ww = p % WP;
  if (hh == 0 || hh == HP - 1 || ww == 0 || ww == WP - 1) {
    float4 z = make_float4(0.f, 0.f, 0.f, 0.f);
    float* k = Kp + (size_t)idx * CH;
    float* v = Vp + (size_t)idx * CH;
#pragma unroll
    for (int i = 0; i < CH; i += 4) {
      *(float4*)(k + i) = z;
      *(float4*)(v + i) = z;
    }
  }
}

// ---------------------------------------------------------------------------
// Kernel 1: fused QKV GEMM.  One block = one image row (128 pixels) of one
// batch.  x tile (64ch x 128px) converted to bf16 in LDS (pixel-major), three
// 64x64 weight matrices converted to bf16 in LDS (row-major).  Each of the 8
// waves owns 16 pixels and runs 3 mats x 4 o-tiles x 2 k-steps = 24 WMMAs.
//   D[16px x 16o] = A[16px x 32c] * B[32c x 16o],  B = W^T
// Outputs: Q in NHWC [B][H][W][64], K/V in padded NHWC [B][130][130][64].
// ---------------------------------------------------------------------------
__global__ __launch_bounds__(256) void SA2D_qkv_gemm(
    const float* __restrict__ x,
    const float* __restrict__ key_w, const float* __restrict__ query_w,
    const float* __restrict__ value_w,
    float* __restrict__ Q, float* __restrict__ Kp, float* __restrict__ Vp) {
  __shared__ __bf16 wlds[3][CH][CH];   // bf16 weights, row-major [o][c]
  __shared__ __bf16 xs[WW][72];        // pixel-major x tile, padded stride

  const int tid = threadIdx.x;
  const int t   = blockIdx.x;          // 0 .. B*H-1
  const int b   = t >> 7;
  const int h   = t & (HH - 1);

  // --- stage weights (f32 -> bf16) ---
  const float* wsrc[3] = {key_w, query_w, value_w};
#pragma unroll
  for (int m = 0; m < 3; ++m)
    for (int i = tid; i < CH * CH; i += 256)
      ((__bf16*)wlds[m])[i] = (__bf16)wsrc[m][i];

  // --- stage x row tile: x[b][c][h][0..127] -> xs[pixel][c] (bf16) ---
  {
    const int c  = tid >> 2;            // 64 channels
    const int p0 = (tid & 3) * 32;      // 4 pixel chunks of 32
    const float* src = x + (((size_t)b * CH + c) * HH + h) * WW + p0;
#pragma unroll
    for (int i = 0; i < 32; i += 4) {
      float4 f = *(const float4*)(src + i);
      xs[p0 + i + 0][c] = (__bf16)f.x;
      xs[p0 + i + 1][c] = (__bf16)f.y;
      xs[p0 + i + 2][c] = (__bf16)f.z;
      xs[p0 + i + 3][c] = (__bf16)f.w;
    }
  }
  __syncthreads();

  const int wave  = tid >> 5;
  const int lane  = tid & 31;
  const int half  = lane >> 4;     // lane half (ISA fragment layout)
  const int ln    = lane & 15;
  const int ptile = wave * 16;     // this wave's 16 pixels (w coordinate base)

  // --- A fragments (16px x 32c), two k-steps, shared by all 12 (mat,o) tiles.
  // Element e of the v16bf: K = half*16 + e  (contiguous, matches B)
  v16bf a[2];
  {
    const int p = ptile + ln;       // pixel row M = lane&15
#pragma unroll
    for (int ks = 0; ks < 2; ++ks) {
      const v8bf lo = *(const v8bf*)&xs[p][ks * 32 + half * 16];
      const v8bf hi = *(const v8bf*)&xs[p][ks * 32 + half * 16 + 8];
#pragma unroll
      for (int e = 0; e < 8; ++e) { a[ks][e] = lo[e]; a[ks][8 + e] = hi[e]; }
    }
  }

#pragma unroll
  for (int m = 0; m < 3; ++m) {
#pragma unroll
    for (int ot = 0; ot < 4; ++ot) {
      v8f acc = {};
      const int o = ot * 16 + ln;   // B column N = lane&15 -> output channel
#pragma unroll
      for (int ks = 0; ks < 2; ++ks) {
        // B fragment: 16 contiguous bf16 of weight row o (K = ks*32+half*16+e)
        v16bf bf = *(const v16bf*)&wlds[m][o][ks * 32 + half * 16];
        acc = __builtin_amdgcn_wmma_f32_16x16x32_bf16(
            /*neg_a=*/false, a[ks], /*neg_b=*/false, bf,
            /*c_mod=*/(short)0, acc, /*reuse_a=*/false, /*reuse_b=*/false);
      }
      // D: VGPR r -> pixel (r + 8*half), col o.  Coalesced 64B half-wave rows.
#pragma unroll
      for (int r = 0; r < 8; ++r) {
        const int pw = ptile + r + half * 8;   // w coordinate
        if (m == 1) {
          Q[((((size_t)b * HH + h) * WW) + pw) * CH + o] = acc[r];
        } else {
          float* outb = (m == 0) ? Kp : Vp;
          outb[((((size_t)b * HP + (h + 1)) * WP) + (pw + 1)) * CH + o] = acc[r];
        }
      }
    }
  }
}

// ---------------------------------------------------------------------------
// Kernel 2: windowed attention.  One block = 32 consecutive pixels of one
// image row.  The 3x34-column K and V neighborhoods (2 x 26 KB) are staged to
// LDS — via GLOBAL_LOAD_ASYNC_TO_LDS_B128 when available — guaranteeing the
// up-to-9x K/V reuse on-chip.  8 lanes per pixel, each lane owns 8 channels;
// full 64-wide dots via 3-step shfl_xor tree; softmax over 9 in registers.
// ---------------------------------------------------------------------------
__global__ __launch_bounds__(256) void SA2D_attn(
    const float* __restrict__ Q, const float* __restrict__ Kp,
    const float* __restrict__ Vp, float* __restrict__ out) {
  __shared__ float kt[3][34][CH];   // padded rows h..h+2, cols w0..w0+33
  __shared__ float vt[3][34][CH];

  const int tid   = threadIdx.x;
  const int gbase = blockIdx.x * 32;            // first pixel of this block
  const int b  = gbase >> 14;
  const int lb = gbase & 16383;
  const int h  = lb >> 7;
  const int w0 = lb & 127;                      // multiple of 32

  // --- stage K/V tiles: 3*34 vectors of 64 floats = 1632 float4 each ---
  const size_t gk = (((size_t)b * HP + h) * WP + w0) * CH;
  for (int i = tid; i < 3 * 34 * 16; i += 256) {
    const int r   = i / (34 * 16);
    const int rem = i - r * (34 * 16);
    const int c   = rem >> 4;                   // tile column 0..33
    const int q4  = (rem & 15) * 4;             // float4 chunk within vector
    const size_t goff = gk + ((size_t)r * WP + c) * CH + q4;
#if SA2D_ASYNC_LDS
    __builtin_amdgcn_global_load_async_to_lds_b128(
        (AS1 v4i*)(Kp + goff), (AS3 v4i*)&kt[r][c][q4], 0, 0);
    __builtin_amdgcn_global_load_async_to_lds_b128(
        (AS1 v4i*)(Vp + goff), (AS3 v4i*)&vt[r][c][q4], 0, 0);
#else
    *(float4*)&kt[r][c][q4] = *(const float4*)(Kp + goff);
    *(float4*)&vt[r][c][q4] = *(const float4*)(Vp + goff);
#endif
  }

  // Q has no reuse: straight global loads overlapping the async staging.
  const int wl    = tid >> 3;                   // local pixel 0..31
  const int cbase = (tid & 7) * 8;              // this lane's 8 channels
  const int gpix  = gbase + wl;
  const float* qv = Q + (size_t)gpix * CH + cbase;
  float4 q0 = *(const float4*)qv, q1 = *(const float4*)(qv + 4);

#if SA2D_ASYNC_LDS
  __builtin_amdgcn_s_wait_asynccnt(0);
#endif
  __syncthreads();

  float s[9];
#pragma unroll
  for (int kk = 0; kk < 9; ++kk) {
    const int dh = kk / 3, dw = kk % 3;         // tile coords: row dh, col wl+dw
    const float* kv = &kt[dh][wl + dw][cbase];
    float4 k0 = *(const float4*)kv, k1 = *(const float4*)(kv + 4);
    float d = q0.x * k0.x + q0.y * k0.y + q0.z * k0.z + q0.w * k0.w +
              q1.x * k1.x + q1.y * k1.y + q1.z * k1.z + q1.w * k1.w;
    d += __shfl_xor(d, 1, 8);
    d += __shfl_xor(d, 2, 8);
    d += __shfl_xor(d, 4, 8);
    s[kk] = d;                                  // full dot in every lane
  }
  float mx = s[0];
#pragma unroll
  for (int kk = 1; kk < 9; ++kk) mx = fmaxf(mx, s[kk]);
  float sum = 0.f;
#pragma unroll
  for (int kk = 0; kk < 9; ++kk) { s[kk] = __expf(s[kk] - mx); sum += s[kk]; }
  const float inv = 1.0f / sum;

  float4 o0 = make_float4(0.f, 0.f, 0.f, 0.f), o1 = o0;
#pragma unroll
  for (int kk = 0; kk < 9; ++kk) {
    const int dh = kk / 3, dw = kk % 3;
    const float* vv = &vt[dh][wl + dw][cbase];
    float4 v0 = *(const float4*)vv, v1 = *(const float4*)(vv + 4);
    const float aw = s[kk] * inv;
    o0.x += aw * v0.x; o0.y += aw * v0.y; o0.z += aw * v0.z; o0.w += aw * v0.w;
    o1.x += aw * v1.x; o1.y += aw * v1.y; o1.z += aw * v1.z; o1.w += aw * v1.w;
  }
  float* op = out + (size_t)gpix * CH + cbase;
  *(float4*)op       = o0;
  *(float4*)(op + 4) = o1;
}

// ---------------------------------------------------------------------------
extern "C" void kernel_launch(void* const* d_in, const int* in_sizes, int n_in,
                              void* d_out, int out_size, void* d_ws,
                              size_t ws_size, hipStream_t stream) {
  (void)in_sizes; (void)n_in; (void)out_size; (void)ws_size;
  const float* x       = (const float*)d_in[0];
  const float* key_w   = (const float*)d_in[1];
  const float* query_w = (const float*)d_in[2];
  const float* value_w = (const float*)d_in[3];
  float* out = (float*)d_out;

  float* Q  = (float*)d_ws;                               // 8*128*128*64 f32
  float* Kp = Q  + (size_t)BATCH * HH * WW * CH;          // 8*130*130*64 f32
  float* Vp = Kp + (size_t)BATCH * HP * WP * CH;          // 8*130*130*64 f32

  SA2D_zero_pad<<<(BATCH * HP * WP + 255) / 256, 256, 0, stream>>>(Kp, Vp);
  SA2D_qkv_gemm<<<BATCH * HH, 256, 0, stream>>>(x, key_w, query_w, value_w,
                                                Q, Kp, Vp);
  SA2D_attn<<<BATCH * HH * WW / 32, 256, 0, stream>>>(Q, Kp, Vp, out);
}
